// NetworkliGNN_54769422958787
// MI455X (gfx1250) — matrix operand order repaired and checked by reading, not compiled, not run on hardware
//
#include <hip/hip_runtime.h>
#include <hip/hip_bf16.h>

// ---------------------------------------------------------------------------
// Problem constants (match reference)
// ---------------------------------------------------------------------------
#define NN    10000
#define EE    80000
#define INF_  256
#define TT    768
#define H1_   1024
#define H2_   1024
#define HEADS 4
#define HC    256      // H2/HEADS
#define NDm   100
#define NSk   10000
#define SDk   128
#define BB    1024
#define KK    20
#define CWID  3328     // IN + 4*T
#define LN_EPS 1e-5f
#define COS_EPS 1e-8f
#define NEG_SLOPE 0.2f

typedef __attribute__((ext_vector_type(16))) __bf16 v16bf;
typedef __attribute__((ext_vector_type(2)))  __bf16 v2bf;
typedef __attribute__((ext_vector_type(8)))  float  v8f;

// canonical shape for clang to emit v_cvt_pk_bf16_f32 vD, vA, vB
__device__ __forceinline__ unsigned pack_bf16(float a, float b) {
    v2bf t;
    t[0] = (__bf16)a;
    t[1] = (__bf16)b;
    unsigned u;
    __builtin_memcpy(&u, &t, 4);
    return u;
}

// ---------------------------------------------------------------------------
// WMMA tiled GEMM:  C[M,Nc] = A[M,K] @ Bw[K,Nc] (+bias) (opt relu)
// Block tile 128x128, 8 waves (2 in M x 4 in N), wave tile 64x32 = 4x2 WMMA
// 16x16x32 bf16.  LDS tiles stored PRE-SWIZZLED in the WMMA fragment layout
// (fragment load = one contiguous 32B read per lane -> 2x ds_load_b128).
// Double-buffered LDS: next tile's global loads are issued before the WMMAs
// so HBM latency hides under the matrix pipe; branch-free staging (row clamp
// + zero mask) lets the compiler batch all staging loads behind one wait.
// ---------------------------------------------------------------------------
#define TBM 128
#define TBN 128
#define TBK 32

__global__ __launch_bounds__(256) void gemm_bias_kernel(
    const float* __restrict__ A, int lda,
    const float* __restrict__ Bw,            // K x Nc row-major
    const float* __restrict__ bias,          // may be nullptr
    float* __restrict__ C, int ldc, int colOff,
    int M, int K, int Nc, int relu)
{
    // [buffer][subtile][lane][dword] ; 8 KB per tile per buffer
    __shared__ unsigned AsP[2][8][32][8];
    __shared__ unsigned BsP[2][8][32][8];

    const int tid  = threadIdx.x;
    const int lane = tid & 31;
    const int wave = tid >> 5;
    const int wm   = wave >> 2;          // 0..1  (M direction, 64 rows each)
    const int wn   = wave & 3;           // 0..3  (N direction, 32 cols each)
    const int m0   = blockIdx.x * TBM;
    const int n0   = blockIdx.y * TBN;

    const int r16 = lane & 15;
    const int hk  = lane >> 4;

    v8f acc[4][2] = {};

    // staging registers for one tile (A: 4 float4, B: 2x2 float4)
    float4 aR[4];
    float4 bR[2][2];

    // A staging geometry per task p: id = tid + p*256 -> m (0..127), c (0..7)
    const int aM[4] = { (tid + 0*256) >> 3, (tid + 1*256) >> 3,
                        (tid + 2*256) >> 3, (tid + 3*256) >> 3 };
    const int aC    = tid & 7;
    // B staging geometry per task p: id = tid + p*256 -> kp (0..15), cg (0..31)
    const int bKp[2] = { (tid + 0*256) >> 5, (tid + 1*256) >> 5 };
    const int bCg    = tid & 31;

    // issue global loads for tile at k-offset kk (branch-free, clamped rows)
    auto stage = [&](int kk) {
        #pragma unroll
        for (int p = 0; p < 4; ++p) {
            int gm = m0 + aM[p];
            float msk = (gm < M) ? 1.f : 0.f;
            if (gm >= M) gm = M - 1;                   // always a valid address
            float4 v = *(const float4*)(A + (long)gm * lda + kk + aC * 4);
            aR[p] = make_float4(v.x * msk, v.y * msk, v.z * msk, v.w * msk);
        }
        #pragma unroll
        for (int p = 0; p < 2; ++p) {
            const float* b0 = Bw + (long)(kk + 2 * bKp[p]) * Nc + n0 + bCg * 4;
            bR[p][0] = *(const float4*)b0;
            bR[p][1] = *(const float4*)(b0 + Nc);
        }
    };
    // convert + store staged registers into LDS buffer `buf`
    auto commit = [&](int buf) {
        #pragma unroll
        for (int p = 0; p < 4; ++p) {
            int m = aM[p], c = aC;
            uint2 pr;
            pr.x = pack_bf16(aR[p].x, aR[p].y);
            pr.y = pack_bf16(aR[p].z, aR[p].w);
            int h   = c >> 2;
            int dwb = 2 * (c & 3);                     // even -> b64 store
            int la  = (m & 15) + 16 * h;
            *(uint2*)&AsP[buf][m >> 4][la][dwb] = pr;
        }
        #pragma unroll
        for (int p = 0; p < 2; ++p) {
            int kp = bKp[p], cg = bCg;
            int sn = cg >> 2;
            int lb = (cg & 3) * 4 + 16 * (kp >> 3);
            int dw = kp & 7;
            BsP[buf][sn][lb + 0][dw] = pack_bf16(bR[p][0].x, bR[p][1].x);
            BsP[buf][sn][lb + 1][dw] = pack_bf16(bR[p][0].y, bR[p][1].y);
            BsP[buf][sn][lb + 2][dw] = pack_bf16(bR[p][0].z, bR[p][1].z);
            BsP[buf][sn][lb + 3][dw] = pack_bf16(bR[p][0].w, bR[p][1].w);
        }
    };

    // prologue: stage tile 0
    stage(0);
    commit(0);
    int cur = 0;

    for (int k0 = 0; k0 < K; k0 += TBK) {
        __syncthreads();                    // buf[cur] ready; buf[cur^1] free
        const bool more = (k0 + TBK < K);
        if (more) stage(k0 + TBK);          // loads in flight during WMMAs
        if (k0 + 2 * TBK < K) {             // warm L2 two tiles ahead
            int gm = m0 + aM[0]; if (gm >= M) gm = M - 1;
            __builtin_prefetch(A  + (long)gm * lda + k0 + 2 * TBK + aC * 4, 0, 1);
            __builtin_prefetch(Bw + (long)(k0 + 2 * TBK + 2 * bKp[0]) * Nc + n0 + bCg * 4, 0, 1);
        }

        // ---- fragment loads: one 32B contiguous LDS read per lane ----
        v16bf afrag[4], bfrag[2];
        #pragma unroll
        for (int i = 0; i < 4; ++i)
            afrag[i] = *(const v16bf*)&AsP[cur][wm * 4 + i][lane][0];
        #pragma unroll
        for (int j = 0; j < 2; ++j)
            bfrag[j] = *(const v16bf*)&BsP[cur][wn * 2 + j][lane][0];

        #pragma unroll
        for (int i = 0; i < 4; ++i)
            #pragma unroll
            for (int j = 0; j < 2; ++j)
                acc[i][j] = __builtin_amdgcn_wmma_f32_16x16x32_bf16(
                    false, afrag[i], false, bfrag[j], (short)0, acc[i][j], false, false);

        if (more) commit(cur ^ 1);          // waits for loads, cvt, ds_store
        cur ^= 1;
    }

    // ---- store C (32-bit C/D layout: VGPR r -> M = 8*half + r, N = lane&15) ----
    #pragma unroll
    for (int i = 0; i < 4; ++i) {
        #pragma unroll
        for (int j = 0; j < 2; ++j) {
            int cn = n0 + (wn * 2 + j) * 16 + r16;
            float bval = bias ? bias[cn] : 0.f;
            #pragma unroll
            for (int r = 0; r < 8; ++r) {
                int cm = m0 + (wm * 4 + i) * 16 + hk * 8 + r;
                if (cm < M) {
                    float v = acc[i][j][r] + bval;
                    if (relu) v = fmaxf(v, 0.f);
                    C[(long)cm * ldc + colOff + cn] = v;
                }
            }
        }
    }
}

// ---------------------------------------------------------------------------
// Small helpers
// ---------------------------------------------------------------------------
__device__ __forceinline__ float wave_sum(float v) {
    #pragma unroll
    for (int o = 16; o > 0; o >>= 1) v += __shfl_xor(v, o, 32);
    return v;
}
__device__ __forceinline__ unsigned enc_f(float f) {
    unsigned u = __float_as_uint(f);
    return (u & 0x80000000u) ? ~u : (u | 0x80000000u);
}
__device__ __forceinline__ float dec_f(unsigned u) {
    return (u & 0x80000000u) ? __uint_as_float(u & 0x7fffffffu) : __uint_as_float(~u);
}

__global__ void copy_x_kernel(const float* __restrict__ x, float* __restrict__ combined) {
    long t = (long)blockIdx.x * blockDim.x + threadIdx.x;
    if (t >= (long)NN * INF_) return;
    long n = t / INF_, c = t % INF_;
    combined[n * CWID + c] = x[t];
}

__global__ void fill_kernel(float* p, long n, float v) {
    long t = (long)blockIdx.x * blockDim.x + threadIdx.x;
    if (t < n) p[t] = v;
}

__global__ void deg_accum_kernel(const int* __restrict__ dstE, float* deg) {
    int t = blockIdx.x * blockDim.x + threadIdx.x;
    if (t < EE) atomicAdd(&deg[dstE[t]], 1.0f);
}

__global__ void finalize_dinv_kernel(float* dinv) {
    int t = blockIdx.x * blockDim.x + threadIdx.x;
    if (t < NN) dinv[t] = rsqrtf(fmaxf(dinv[t], 1.0f));
}

__global__ void init_rows_bias_kernel(float* __restrict__ out, const float* __restrict__ b) {
    long t = (long)blockIdx.x * blockDim.x + threadIdx.x;
    if (t >= (long)NN * H2_) return;
    out[t] = b[t & (H2_ - 1)];
}

// one block per (edge or self-loop); 256 threads * 4 cols
__global__ __launch_bounds__(256) void gcn_scatter_kernel(
    const int* __restrict__ srcE, const int* __restrict__ dstE,
    const float* __restrict__ dinv, const float* __restrict__ xw, float* __restrict__ out)
{
    int eid = blockIdx.x;
    int s = (eid < EE) ? srcE[eid] : eid - EE;
    int d = (eid < EE) ? dstE[eid] : eid - EE;
    float norm = dinv[s] * dinv[d];
    int c = threadIdx.x * 4;
    const float* xp = xw + (long)s * H2_ + c;
    float* op = out + (long)d * H2_ + c;
    #pragma unroll
    for (int k = 0; k < 4; ++k) atomicAdd(op + k, norm * xp[k]);
}

// one wave per (node, head)
__global__ __launch_bounds__(256) void gat_scores_kernel(
    const float* __restrict__ h, const float* __restrict__ a_s, const float* __restrict__ a_d,
    float* __restrict__ es, float* __restrict__ ed)
{
    int gw = (blockIdx.x * blockDim.x + threadIdx.x) >> 5;
    int lane = threadIdx.x & 31;
    if (gw >= NN * HEADS) return;
    int node = gw >> 2, hd = gw & 3;
    const float* hp  = h + (long)node * H2_ + hd * HC;
    const float* asp = a_s + hd * HC;
    const float* adp = a_d + hd * HC;
    float ss = 0.f, sd = 0.f;
    for (int c = lane; c < HC; c += 32) { float v = hp[c]; ss += v * asp[c]; sd += v * adp[c]; }
    ss = wave_sum(ss); sd = wave_sum(sd);
    if (lane == 0) { es[gw] = ss; ed[gw] = sd; }
}

__global__ void gat_init_kernel(float* att, float* ssum, unsigned* mx) {
    long t = (long)blockIdx.x * blockDim.x + threadIdx.x;
    if (t < (long)NN * H2_) att[t] = 0.f;
    if (t < (long)NN * HEADS) { ssum[t] = 0.f; mx[t] = 0u; }
}

__global__ void gat_edge_max_kernel(const int* __restrict__ srcE, const int* __restrict__ dstE,
                                    const float* __restrict__ es, const float* __restrict__ ed,
                                    unsigned* __restrict__ mx)
{
    int t = blockIdx.x * blockDim.x + threadIdx.x;
    if (t >= (EE + NN) * HEADS) return;
    int eid = t >> 2, hd = t & 3;
    int s = (eid < EE) ? srcE[eid] : eid - EE;
    int d = (eid < EE) ? dstE[eid] : eid - EE;
    float e = es[s * HEADS + hd] + ed[d * HEADS + hd];
    e = (e > 0.f) ? e : e * NEG_SLOPE;
    atomicMax(&mx[d * HEADS + hd], enc_f(e));
}

__global__ void gat_edge_sum_kernel(const int* __restrict__ srcE, const int* __restrict__ dstE,
                                    const float* __restrict__ es, const float* __restrict__ ed,
                                    const unsigned* __restrict__ mx, float* __restrict__ ssum)
{
    int t = blockIdx.x * blockDim.x + threadIdx.x;
    if (t >= (EE + NN) * HEADS) return;
    int eid = t >> 2, hd = t & 3;
    int s = (eid < EE) ? srcE[eid] : eid - EE;
    int d = (eid < EE) ? dstE[eid] : eid - EE;
    float e = es[s * HEADS + hd] + ed[d * HEADS + hd];
    e = (e > 0.f) ? e : e * NEG_SLOPE;
    float ee = __expf(e - dec_f(mx[d * HEADS + hd]));
    atomicAdd(&ssum[d * HEADS + hd], ee);
}

__global__ __launch_bounds__(256) void gat_scatter_kernel(
    const int* __restrict__ srcE, const int* __restrict__ dstE,
    const float* __restrict__ es, const float* __restrict__ ed,
    const unsigned* __restrict__ mx, const float* __restrict__ ssum,
    const float* __restrict__ h, float* __restrict__ out)
{
    int eid = blockIdx.x;
    int s = (eid < EE) ? srcE[eid] : eid - EE;
    int d = (eid < EE) ? dstE[eid] : eid - EE;
    int c = threadIdx.x * 4;
    int hd = c >> 8;                 // 256 channels per head
    float e = es[s * HEADS + hd] + ed[d * HEADS + hd];
    e = (e > 0.f) ? e : e * NEG_SLOPE;
    float alpha = __expf(e - dec_f(mx[d * HEADS + hd])) / ssum[d * HEADS + hd];
    const float* hs = h + (long)s * H2_ + c;
    float* op = out + (long)d * H2_ + c;
    #pragma unroll
    for (int k = 0; k < 4; ++k) atomicAdd(op + k, alpha * hs[k]);
}

// v = att + gat_b + gcn; out = relu(layernorm(v) * g + b). One block per row.
__global__ __launch_bounds__(256) void ln_relu_kernel(
    const float* __restrict__ att, const float* __restrict__ gat_b,
    const float* __restrict__ gcn, const float* __restrict__ g,
    const float* __restrict__ b, float* __restrict__ out)
{
    int node = blockIdx.x;
    float v[4], s = 0.f, s2 = 0.f;
    #pragma unroll
    for (int k = 0; k < 4; ++k) {
        int c = threadIdx.x + k * 256;
        float x = att[(long)node * H2_ + c] + gat_b[c] + gcn[(long)node * H2_ + c];
        v[k] = x; s += x; s2 += x * x;
    }
    __shared__ float rs[8], rs2[8];
    s = wave_sum(s); s2 = wave_sum(s2);
    int wave = threadIdx.x >> 5, lane = threadIdx.x & 31;
    if (lane == 0) { rs[wave] = s; rs2[wave] = s2; }
    __syncthreads();
    __shared__ float mean_sh, rinv_sh;
    if (threadIdx.x == 0) {
        float S = 0.f, S2 = 0.f;
        #pragma unroll
        for (int w = 0; w < 8; ++w) { S += rs[w]; S2 += rs2[w]; }
        float m = S / (float)H2_;
        float var = S2 / (float)H2_ - m * m;
        mean_sh = m; rinv_sh = rsqrtf(var + LN_EPS);
    }
    __syncthreads();
    float m = mean_sh, ri = rinv_sh;
    #pragma unroll
    for (int k = 0; k < 4; ++k) {
        int c = threadIdx.x + k * 256;
        float y = (v[k] - m) * ri * g[c] + b[c];
        out[(long)node * H2_ + c] = fmaxf(y, 0.f);
    }
}

__global__ void dom_add_kernel(float* __restrict__ out, const float* __restrict__ dom_emb,
                               const int* __restrict__ domain_ids)
{
    long t = (long)blockIdx.x * blockDim.x + threadIdx.x;
    if (t >= (long)NN * H2_) return;
    long n = t >> 10, c = t & 1023;
    out[t] += dom_emb[(long)domain_ids[n] * H2_ + c];
}

__global__ void skill_mean_kernel(const int* __restrict__ sid1, const int* __restrict__ sid2,
                                  const float* __restrict__ skill_emb, float* __restrict__ comb)
{
    long t = (long)blockIdx.x * blockDim.x + threadIdx.x;
    if (t >= (long)BB * 2 * SDk) return;
    int b = (int)(t >> 8);
    int c = (int)(t & 255);
    const int* ids = (c < SDk) ? (sid1 + b * KK) : (sid2 + b * KK);
    int cc = (c < SDk) ? c : c - SDk;
    float s = 0.f;
    #pragma unroll
    for (int k = 0; k < KK; ++k) s += skill_emb[(long)ids[k] * SDk + cc];
    comb[t] = s / (float)KK;
}

// one wave per batch element: sigmoid(hidden . c_w2 + c_b2)
__global__ __launch_bounds__(256) void compat_out_kernel(
    const float* __restrict__ hidden, const float* __restrict__ c_w2,
    const float* __restrict__ c_b2, float* __restrict__ out)
{
    int gw = (blockIdx.x * blockDim.x + threadIdx.x) >> 5;
    int lane = threadIdx.x & 31;
    if (gw >= BB) return;
    const float* hp = hidden + (long)gw * H2_;
    float s = 0.f;
    for (int c = lane; c < H2_; c += 32) s += hp[c] * c_w2[c];
    s = wave_sum(s);
    if (lane == 0) out[gw] = 1.f / (1.f + __expf(-(s + c_b2[0])));
}

// one wave per (node, feature): norm of combined[n, 256 + f*768 : +768]
__global__ __launch_bounds__(256) void feat_norms_kernel(const float* __restrict__ combined,
                                                         float* __restrict__ fnorm)
{
    int gw = (blockIdx.x * blockDim.x + threadIdx.x) >> 5;
    int lane = threadIdx.x & 31;
    if (gw >= NN * 4) return;
    int node = gw >> 2, f = gw & 3;
    const float* p = combined + (long)node * CWID + INF_ + f * TT;
    float s = 0.f;
    for (int c = lane; c < TT; c += 32) { float v = p[c]; s += v * v; }
    s = wave_sum(s);
    if (lane == 0) fnorm[gw] = fmaxf(sqrtf(s), COS_EPS);
}

// one wave per edge: 4 cosine sims
__global__ __launch_bounds__(256) void edge_sims_kernel(
    const int* __restrict__ srcE, const int* __restrict__ dstE,
    const float* __restrict__ combined, const float* __restrict__ fnorm,
    float* __restrict__ o_bio, float* __restrict__ o_en,
    float* __restrict__ o_ne, float* __restrict__ o_goals)
{
    int gw = (blockIdx.x * blockDim.x + threadIdx.x) >> 5;
    int lane = threadIdx.x & 31;
    if (gw >= EE) return;
    int s = srcE[gw], d = dstE[gw];
    const float* ps = combined + (long)s * CWID + INF_;   // [bio|exp|needs|goals] each 768
    const float* pd = combined + (long)d * CWID + INF_;
    float d0 = 0.f, d1 = 0.f, d2 = 0.f, d3 = 0.f;
    for (int c = lane; c < TT; c += 32) {
        d0 += ps[c]            * pd[c];              // bio . bio
        d1 += ps[TT + c]       * pd[2 * TT + c];     // exp_s . needs_d
        d2 += ps[2 * TT + c]   * pd[TT + c];         // needs_s . exp_d
        d3 += ps[3 * TT + c]   * pd[3 * TT + c];     // goals . goals
    }
    d0 = wave_sum(d0); d1 = wave_sum(d1); d2 = wave_sum(d2); d3 = wave_sum(d3);
    if (lane == 0) {
        o_bio[gw]   = d0 / (fnorm[s * 4 + 0] * fnorm[d * 4 + 0]);
        o_en[gw]    = d1 / (fnorm[s * 4 + 1] * fnorm[d * 4 + 2]);
        o_ne[gw]    = d2 / (fnorm[s * 4 + 2] * fnorm[d * 4 + 1]);
        o_goals[gw] = d3 / (fnorm[s * 4 + 3] * fnorm[d * 4 + 3]);
    }
}

// ---------------------------------------------------------------------------
// Host launch
// ---------------------------------------------------------------------------
static inline int cdiv(long a, long b) { return (int)((a + b - 1) / b); }

extern "C" void kernel_launch(void* const* d_in, const int* in_sizes, int n_in,
                              void* d_out, int out_size, void* d_ws, size_t ws_size,
                              hipStream_t stream) {
    const float* x          = (const float*)d_in[0];
    const int*   ei         = (const int*)d_in[1];
    const int*   domain_ids = (const int*)d_in[2];
    const int*   sid1       = (const int*)d_in[3];
    const int*   sid2       = (const int*)d_in[4];
    const float* feat_emb[4] = { (const float*)d_in[5], (const float*)d_in[8],
                                 (const float*)d_in[11], (const float*)d_in[14] };
    const float* feat_w[4]   = { (const float*)d_in[6], (const float*)d_in[9],
                                 (const float*)d_in[12], (const float*)d_in[15] };
    const float* feat_b[4]   = { (const float*)d_in[7], (const float*)d_in[10],
                                 (const float*)d_in[13], (const float*)d_in[16] };
    const float* gcn_w[2] = { (const float*)d_in[17], (const float*)d_in[25] };
    const float* gcn_b[2] = { (const float*)d_in[18], (const float*)d_in[26] };
    const float* gat_w[2] = { (const float*)d_in[19], (const float*)d_in[27] };
    const float* gat_as[2]= { (const float*)d_in[20], (const float*)d_in[28] };
    const float* gat_ad[2]= { (const float*)d_in[21], (const float*)d_in[29] };
    const float* gat_b[2] = { (const float*)d_in[22], (const float*)d_in[30] };
    const float* ln_g[2]  = { (const float*)d_in[23], (const float*)d_in[31] };
    const float* ln_b[2]  = { (const float*)d_in[24], (const float*)d_in[32] };
    const float* dom_emb  = (const float*)d_in[33];
    const float* dom_w    = (const float*)d_in[34];
    const float* dom_b    = (const float*)d_in[35];
    const float* skill_emb= (const float*)d_in[36];
    const float* c_w1     = (const float*)d_in[37];
    const float* c_b1     = (const float*)d_in[38];
    const float* c_w2     = (const float*)d_in[39];
    const float* c_b2     = (const float*)d_in[40];

    const int* srcE = ei;
    const int* dstE = ei + EE;

    // workspace carve-up (floats)
    float* ws = (float*)d_ws;
    long o = 0;
    float* combined = ws + o;  o += (long)NN * CWID;
    float* xw       = ws + o;  o += (long)NN * H2_;
    float* gcn_out  = ws + o;  o += (long)NN * H2_;
    float* gat_h    = ws + o;  o += (long)NN * H2_;
    float* att      = ws + o;  o += (long)NN * H2_;
    float* hbuf     = ws + o;  o += (long)NN * H2_;
    float* dinv     = ws + o;  o += NN;
    float* es       = ws + o;  o += (long)NN * HEADS;
    float* edv      = ws + o;  o += (long)NN * HEADS;
    float* ssum     = ws + o;  o += (long)NN * HEADS;
    unsigned* mx    = (unsigned*)(ws + o); o += (long)NN * HEADS;
    float* fnorm    = ws + o;  o += (long)NN * HEADS;
    float* comb     = ws + o;  o += (long)BB * 2 * SDk;
    float* hidden   = ws + o;  o += (long)BB * H2_;
    (void)ws_size; (void)in_sizes; (void)n_in; (void)out_size;

    // output carve-up
    float* out        = (float*)d_out;
    float* out_dom    = out;
    float* out_compat = out + (long)NN * H2_;
    float* out_bio    = out_compat + BB;
    float* out_en     = out_bio + EE;
    float* out_ne     = out_en + EE;
    float* out_goals  = out_ne + EE;

    const int gM = cdiv(NN, TBM);  // 79

    // 1) combined[:, :256] = x
    copy_x_kernel<<<cdiv((long)NN * INF_, 256), 256, 0, stream>>>(x, combined);

    // 2) four text projections -> combined[:, 256 + j*768 : ...]
    for (int j = 0; j < 4; ++j) {
        gemm_bias_kernel<<<dim3(gM, TT / TBN), 256, 0, stream>>>(
            feat_emb[j], TT, feat_w[j], feat_b[j],
            combined, CWID, INF_ + j * TT, NN, TT, TT, 0);
    }

    // 3) degree / dinv  (self loop -> start at 1)
    fill_kernel<<<cdiv(NN, 256), 256, 0, stream>>>(dinv, NN, 1.0f);
    deg_accum_kernel<<<cdiv(EE, 256), 256, 0, stream>>>(dstE, dinv);
    finalize_dinv_kernel<<<cdiv(NN, 256), 256, 0, stream>>>(dinv);

    // 4) two GNN layers
    const float* layer_in  = combined;
    int          layer_lda = CWID, layer_K = CWID;
    for (int L = 0; L < 2; ++L) {
        // GCN: xw = in @ gcn_w
        gemm_bias_kernel<<<dim3(gM, H2_ / TBN), 256, 0, stream>>>(
            layer_in, layer_lda, gcn_w[L], nullptr, xw, H2_, 0, NN, layer_K, H2_, 0);
        // gcn_out = bias, then scatter norm * xw[src] into dst
        init_rows_bias_kernel<<<cdiv((long)NN * H2_, 256), 256, 0, stream>>>(gcn_out, gcn_b[L]);
        gcn_scatter_kernel<<<EE + NN, 256, 0, stream>>>(srcE, dstE, dinv, xw, gcn_out);

        // GAT: h = gcn_out @ gat_w (gat bias folded into LN input)
        gemm_bias_kernel<<<dim3(gM, H2_ / TBN), 256, 0, stream>>>(
            gcn_out, H2_, gat_w[L], nullptr, gat_h, H2_, 0, NN, H2_, H2_, 0);
        gat_scores_kernel<<<cdiv((long)NN * HEADS * 32, 256), 256, 0, stream>>>(
            gat_h, gat_as[L], gat_ad[L], es, edv);
        gat_init_kernel<<<cdiv((long)NN * H2_, 256), 256, 0, stream>>>(att, ssum, mx);
        gat_edge_max_kernel<<<cdiv((long)(EE + NN) * HEADS, 256), 256, 0, stream>>>(
            srcE, dstE, es, edv, mx);
        gat_edge_sum_kernel<<<cdiv((long)(EE + NN) * HEADS, 256), 256, 0, stream>>>(
            srcE, dstE, es, edv, mx, ssum);
        gat_scatter_kernel<<<EE + NN, 256, 0, stream>>>(srcE, dstE, es, edv, mx, ssum, gat_h, att);

        // residual + LN + ReLU -> hbuf
        ln_relu_kernel<<<NN, 256, 0, stream>>>(att, gat_b[L], gcn_out, ln_g[L], ln_b[L], hbuf);

        layer_in = hbuf; layer_lda = H2_; layer_K = H2_;
    }

    // 5) domain embeddings = h @ dom_w + dom_b + dom_emb[domain_ids]
    gemm_bias_kernel<<<dim3(gM, H2_ / TBN), 256, 0, stream>>>(
        hbuf, H2_, dom_w, dom_b, out_dom, H2_, 0, NN, H2_, H2_, 0);
    dom_add_kernel<<<cdiv((long)NN * H2_, 256), 256, 0, stream>>>(out_dom, dom_emb, domain_ids);

    // 6) skill compatibility
    skill_mean_kernel<<<cdiv((long)BB * 2 * SDk, 256), 256, 0, stream>>>(sid1, sid2, skill_emb, comb);
    gemm_bias_kernel<<<dim3(cdiv(BB, TBM), H2_ / TBN), 256, 0, stream>>>(
        comb, 2 * SDk, c_w1, c_b1, hidden, H2_, 0, BB, 2 * SDk, H2_, 1);
    compat_out_kernel<<<cdiv((long)BB * 32, 256), 256, 0, stream>>>(hidden, c_w2, c_b2, out_compat);

    // 7) cosine sims over the original E edges
    feat_norms_kernel<<<cdiv((long)NN * 4 * 32, 256), 256, 0, stream>>>(combined, fnorm);
    edge_sims_kernel<<<cdiv((long)EE * 32, 256), 256, 0, stream>>>(
        srcE, dstE, combined, fnorm, out_bio, out_en, out_ne, out_goals);
}
